// HunyuanTopKGate_86775519248867
// MI455X (gfx1250) — compile-verified, matrix operand order checked by reading, not run on hardware
//
#include <hip/hip_runtime.h>
#include <stdint.h>

#define TOKENS 4096
#define HDIM   4096
#define NEXP   64
#define NTOPK  8
#define FP32EPS 1.1920929e-07f

typedef float  f32x2  __attribute__((ext_vector_type(2)));
typedef float  f32x4  __attribute__((ext_vector_type(4)));
typedef float  f32x8  __attribute__((ext_vector_type(8)));
typedef __bf16 bf16x16 __attribute__((ext_vector_type(16)));

#if __has_builtin(__builtin_amdgcn_wmma_f32_16x16x4_f32)
#define USE_WMMA_F32 1
#else
#define USE_WMMA_F32 0
#endif

// ---------------------------------------------------------------------------
// Kernel 1: logits[T, 64] = x[T, 4096] @ W[64, 4096]^T   (fp32 WMMA)
// One wave computes a 16(M) x 64(N) strip: A tile loaded once, reused for
// 4 WMMAs -> x is read from HBM exactly once (64 MB, the roofline limit).
// WMMA f32 16x16x4 layout: lane L -> row/col = L%16, K pair = 2*(L/16)+{0,1}.
// ---------------------------------------------------------------------------
__global__ __launch_bounds__(256) void gate_gemm(const float* __restrict__ x,
                                                 const float* __restrict__ w,
                                                 float* __restrict__ logits) {
  const int lane  = threadIdx.x & 31;
  const int mtile = blockIdx.x * (blockDim.x >> 5) + (threadIdx.x >> 5); // 0..255
  const int m0    = mtile * 16;
  const int nr    = lane & 15;   // A row / B col / C col within tile
  const int h     = lane >> 4;   // half selector

  const float* arow  = x + (size_t)(m0 + nr) * HDIM;
  const float* brow0 = w + (size_t)( 0 + nr) * HDIM;
  const float* brow1 = w + (size_t)(16 + nr) * HDIM;
  const float* brow2 = w + (size_t)(32 + nr) * HDIM;
  const float* brow3 = w + (size_t)(48 + nr) * HDIM;

  f32x8 acc0 = {}, acc1 = {}, acc2 = {}, acc3 = {};

#if USE_WMMA_F32
  #pragma unroll 8
  for (int k = 0; k < HDIM; k += 4) {
    const int ka = k + 2 * h;
    f32x2 a  = *(const f32x2*)(arow  + ka);
    f32x2 b0 = *(const f32x2*)(brow0 + ka);
    f32x2 b1 = *(const f32x2*)(brow1 + ka);
    f32x2 b2 = *(const f32x2*)(brow2 + ka);
    f32x2 b3 = *(const f32x2*)(brow3 + ka);
    acc0 = __builtin_amdgcn_wmma_f32_16x16x4_f32(false, a, false, b0, (short)0, acc0, false, false);
    acc1 = __builtin_amdgcn_wmma_f32_16x16x4_f32(false, a, false, b1, (short)0, acc1, false, false);
    acc2 = __builtin_amdgcn_wmma_f32_16x16x4_f32(false, a, false, b2, (short)0, acc2, false, false);
    acc3 = __builtin_amdgcn_wmma_f32_16x16x4_f32(false, a, false, b3, (short)0, acc3, false, false);
  }
#else
  // bf16 WMMA fallback (codegen-confirmed builtin), K-step 32.
  // 16-bit A 16x32 layout: lanes<16 hold K={0..7,16..23}, lanes>=16 K={8..15,24..31}.
  for (int k = 0; k < HDIM; k += 32) {
    const int ka = k + 8 * h;
    bf16x16 a, b0, b1, b2, b3;
    #pragma unroll
    for (int j = 0; j < 8; ++j) {
      a[j]  = (__bf16)arow[ka + j];       a[j + 8]  = (__bf16)arow[ka + 16 + j];
      b0[j] = (__bf16)brow0[ka + j];      b0[j + 8] = (__bf16)brow0[ka + 16 + j];
      b1[j] = (__bf16)brow1[ka + j];      b1[j + 8] = (__bf16)brow1[ka + 16 + j];
      b2[j] = (__bf16)brow2[ka + j];      b2[j + 8] = (__bf16)brow2[ka + 16 + j];
      b3[j] = (__bf16)brow3[ka + j];      b3[j + 8] = (__bf16)brow3[ka + 16 + j];
    }
    acc0 = __builtin_amdgcn_wmma_f32_16x16x32_bf16(false, a, false, b0, (short)0, acc0, false, false);
    acc1 = __builtin_amdgcn_wmma_f32_16x16x32_bf16(false, a, false, b1, (short)0, acc1, false, false);
    acc2 = __builtin_amdgcn_wmma_f32_16x16x32_bf16(false, a, false, b2, (short)0, acc2, false, false);
    acc3 = __builtin_amdgcn_wmma_f32_16x16x32_bf16(false, a, false, b3, (short)0, acc3, false, false);
  }
#endif

  // C/D layout: VGPR v -> M = m0 + v + 8*(lane/16), N = lane%16 (+ tile base)
  #pragma unroll
  for (int v = 0; v < 8; ++v) {
    float* lr = logits + (size_t)(m0 + v + 8 * h) * NEXP + nr;
    lr[0]  = acc0[v];
    lr[16] = acc1[v];
    lr[32] = acc2[v];
    lr[48] = acc3[v];
  }
}

// ---------------------------------------------------------------------------
// Kernel 2: per-token softmax over 64 experts + iterative top-8 (stable,
// lower-index tie break like jax.lax.top_k). One wave32 per token; each lane
// holds gates for experts {lane, lane+32}. Writes idx[t,k] and gate/denom.
// ---------------------------------------------------------------------------
__global__ __launch_bounds__(256) void topk_softmax(const float* __restrict__ logits,
                                                    int* __restrict__ idx,
                                                    float* __restrict__ wsel) {
  const int t    = blockIdx.x * (blockDim.x >> 5) + (threadIdx.x >> 5);
  const int lane = threadIdx.x & 31;

  float a0 = logits[(size_t)t * NEXP + lane];
  float a1 = logits[(size_t)t * NEXP + 32 + lane];

  float m = fmaxf(a0, a1);
  #pragma unroll
  for (int o = 16; o; o >>= 1) m = fmaxf(m, __shfl_xor(m, o, 32));
  float e0 = __expf(a0 - m), e1 = __expf(a1 - m);
  float s = e0 + e1;
  #pragma unroll
  for (int o = 16; o; o >>= 1) s += __shfl_xor(s, o, 32);
  float c0 = e0 / s, c1 = e1 / s;   // gates

  float sv0, sv1, sv2, sv3, sv4, sv5, sv6, sv7;
  int   si0, si1, si2, si3, si4, si5, si6, si7;
  float selsum = 0.f;
  #pragma unroll
  for (int kk = 0; kk < NTOPK; ++kk) {
    float v; int id;
    if (c0 >= c1) { v = c0; id = lane; } else { v = c1; id = lane + 32; }
    #pragma unroll
    for (int o = 16; o; o >>= 1) {
      float ov = __shfl_xor(v, o, 32);
      int   oi = __shfl_xor(id, o, 32);
      if (ov > v || (ov == v && oi < id)) { v = ov; id = oi; }
    }
    selsum += v;
    if (kk == 0) { sv0 = v; si0 = id; } else if (kk == 1) { sv1 = v; si1 = id; }
    else if (kk == 2) { sv2 = v; si2 = id; } else if (kk == 3) { sv3 = v; si3 = id; }
    else if (kk == 4) { sv4 = v; si4 = id; } else if (kk == 5) { sv5 = v; si5 = id; }
    else if (kk == 6) { sv6 = v; si6 = id; } else { sv7 = v; si7 = id; }
    if (lane == (id & 31)) { if (id < 32) c0 = -1.f; else c1 = -1.f; }  // gates > 0
  }
  const float rdenom = 1.f / fmaxf(selsum, FP32EPS);
  if (lane == 0) {
    int*   ip = idx  + (size_t)t * NTOPK;
    float* wp = wsel + (size_t)t * NTOPK;
    ip[0] = si0; ip[1] = si1; ip[2] = si2; ip[3] = si3;
    ip[4] = si4; ip[5] = si5; ip[6] = si6; ip[7] = si7;
    wp[0] = sv0 * rdenom; wp[1] = sv1 * rdenom; wp[2] = sv2 * rdenom; wp[3] = sv3 * rdenom;
    wp[4] = sv4 * rdenom; wp[5] = sv5 * rdenom; wp[6] = sv6 * rdenom; wp[7] = sv7 * rdenom;
  }
}

// ---------------------------------------------------------------------------
// Kernel 3: deterministic priority. Flattened order p = k*T + t (k-major,
// token-minor, matching idx.T.reshape(-1)). One wave per expert: 32 flat
// slots per step, ballot + prefix-popcount gives stable ranks.
// Since capacity == max expert count, every assignment is within capacity.
// ---------------------------------------------------------------------------
__global__ __launch_bounds__(32) void expert_priority(const int* __restrict__ idx,
                                                      int* __restrict__ prio) {
  const int e    = blockIdx.x;
  const int lane = threadIdx.x;
  int base = 0;
  for (int p0 = 0; p0 < NTOPK * TOKENS; p0 += 32) {
    const int p = p0 + lane;
    const int k = p >> 12;       // p / 4096
    const int t = p & (TOKENS - 1);
    const bool match = (idx[(size_t)t * NTOPK + k] == e);
    const unsigned bm = (unsigned)__ballot(match);
    if (match) prio[(size_t)t * NTOPK + k] = base + __popc(bm & ((1u << lane) - 1u));
    base += __popc(bm);
  }
}

// ---------------------------------------------------------------------------
// Kernel 4: zero-fill d_out (~1.3 GB -> the dominant cost). 16B non-temporal
// stores so the streaming zeros don't thrash L2.
// ---------------------------------------------------------------------------
__global__ __launch_bounds__(256) void zero_out(f32x4* __restrict__ out, size_t n4) {
  size_t i = (size_t)blockIdx.x * blockDim.x + threadIdx.x;
  const size_t stride = (size_t)gridDim.x * blockDim.x;
  const f32x4 z = {0.f, 0.f, 0.f, 0.f};
  for (; i < n4; i += stride) __builtin_nontemporal_store(z, &out[i]);
}

// ---------------------------------------------------------------------------
// Kernel 5: scatter the 32768 nonzero entries.
// combine_weights[t,e,c] = gate/denom ; dispatch_mask[t,e,c] = 1.0
// ---------------------------------------------------------------------------
__global__ __launch_bounds__(256) void scatter(const int* __restrict__ idx,
                                               const int* __restrict__ prio,
                                               const float* __restrict__ wsel,
                                               float* __restrict__ out, int cap) {
  const int i = blockIdx.x * blockDim.x + threadIdx.x;
  if (i >= TOKENS * NTOPK) return;
  const int t = i >> 3;
  const int e = idx[i];
  const int c = prio[i];
  const size_t off = ((size_t)t * NEXP + e) * (size_t)cap + c;
  out[off] = wsel[i];                                   // combine_weights
  out[(size_t)TOKENS * NEXP * cap + off] = 1.0f;        // dispatch_mask
}

// ---------------------------------------------------------------------------
extern "C" void kernel_launch(void* const* d_in, const int* in_sizes, int n_in,
                              void* d_out, int out_size, void* d_ws, size_t ws_size,
                              hipStream_t stream) {
  (void)in_sizes; (void)n_in; (void)ws_size;
  const float* x = (const float*)d_in[0];   // [1,4096,4096] fp32
  const float* w = (const float*)d_in[1];   // [64,4096] fp32
  float* out = (float*)d_out;

  char* ws = (char*)d_ws;
  float* logits = (float*)ws;                                       // 1 MB
  int*   idx    = (int*)  (ws + (size_t)TOKENS * NEXP * 4);         // 128 KB
  float* wsel   = (float*)(ws + (size_t)TOKENS * NEXP * 4
                              + (size_t)TOKENS * NTOPK * 4);        // 128 KB
  int*   prio   = (int*)  (ws + (size_t)TOKENS * NEXP * 4
                              + (size_t)TOKENS * NTOPK * 8);        // 128 KB

  // capacity is encoded in the harness-provided output size:
  // out_size = 2 * T * E * C
  int cap = out_size / (2 * TOKENS * NEXP);
  if (cap < 1) cap = 1;

  gate_gemm<<<TOKENS / (16 * 8), 256, 0, stream>>>(x, w, logits);
  topk_softmax<<<TOKENS / 8, 256, 0, stream>>>(logits, idx, wsel);
  expert_priority<<<NEXP, 32, 0, stream>>>(idx, prio);

  const size_t n4 = (size_t)out_size / 4;   // out_size divisible by 4 (T*E mult of 4)
  zero_out<<<8192, 256, 0, stream>>>((f32x4*)out, n4);
  scatter<<<(TOKENS * NTOPK) / 256, 256, 0, stream>>>(idx, prio, wsel, out, cap);
}
